// MHGAT_22247930594091
// MI455X (gfx1250) — matrix operand order repaired and checked by reading, not compiled, not run on hardware
//
#include <hip/hip_runtime.h>
#include <hip/hip_bf16.h>
#include <cstdint>

// ---------------------------------------------------------------------------
// MHGAT on gfx1250: 4 dense GEMMs via v_wmma_f32_16x16x32_f16 (f16 in, f32
// accum) with async global->LDS staging; graph part via float atomics.
// ---------------------------------------------------------------------------

typedef __attribute__((ext_vector_type(16))) _Float16 v16h;
typedef __attribute__((ext_vector_type(8)))  float    v8f;

// Pointee type the async-LDS builtin expects (from hipcc diagnostic):
//   int __attribute__((vector_size(4*sizeof(int)))) __device__ *
typedef int b128_t __attribute__((vector_size(4 * sizeof(int))));
typedef __attribute__((address_space(1))) b128_t* gptr_b128;
typedef __attribute__((address_space(3))) b128_t* lptr_b128;

// Generic -> AS1: same numeric address on AMDGPU, go via 64-bit integer.
#define TO_GLOBAL_B128(p) ((gptr_b128)(unsigned long long)(p))
// Generic LDS pointer -> AS3: low 32 bits of the generic address are the LDS
// byte address (ISA 10.2 aperture rules); integer->pointer cast is always OK.
#define TO_LDS_B128(p)    ((lptr_b128)(unsigned)(unsigned long long)(p))

#define TM 128
#define TN 128
#define TK 32
#define APAD 12   // fp32 A tile row stride = 44 dwords: rows hit distinct banks
#define BPAD 8    // f16  B tile row stride = 40 halves: rows hit distinct banks

#if defined(__gfx1250__) && \
    __has_builtin(__builtin_amdgcn_global_load_async_to_lds_b128) && \
    __has_builtin(__builtin_amdgcn_s_wait_asynccnt)
#define USE_ASYNC_LDS 1
#else
#define USE_ASYNC_LDS 0
#endif

// ---------------------------------------------------------------------------
// Tiled GEMM: C[M,Ncol] = A[M,K] * B[K,Ncol], fp32 in/out, f16 WMMA compute.
// 256 threads = 8 waves; block tile 128x128; each wave owns a 32x64 sub-tile
// (2x4 WMMA 16x16x32 accumulators -> 8 WMMAs per K-step per wave).
// Requires: Ncol % 128 == 0, K % 4 == 0 (rows 16B-aligned).
// ---------------------------------------------------------------------------
__global__ __launch_bounds__(256) void gemm_wmma_f16(
    const float* __restrict__ A, const float* __restrict__ B,
    float* __restrict__ C, int M, int K, int Ncol)
{
    __shared__ float    As[TM][TK + APAD];   // fp32 staging [m][k] (async copy)
    __shared__ _Float16 Bs[TN][TK + BPAD];   // f16 transposed [n][k]

    const int bm   = blockIdx.y * TM;
    const int bn   = blockIdx.x * TN;
    const int tid  = threadIdx.x;
    const int wave = tid >> 5;
    const int lane = tid & 31;
    const int wm   = (wave >> 1) * 32;       // 0,32,64,96
    const int wn   = (wave & 1) * 64;        // 0,64

    v8f acc[2][4] = {};

    for (int k0 = 0; k0 < K; k0 += TK) {
        // ---- Stage A tile (128 x 32 fp32) in 16B quads: 4 quads/thread. ----
        #pragma unroll
        for (int it = 0; it < (TM * TK / 4) / 256; ++it) {
            int i  = tid + it * 256;
            int r  = i >> 3;                 // 8 quads per row
            int c  = (i & 7) * 4;
            int gr = bm + r, gc = k0 + c;
            if (gr < M && gc + 3 < K) {
#if USE_ASYNC_LDS
                __builtin_amdgcn_global_load_async_to_lds_b128(
                    TO_GLOBAL_B128(A + (size_t)gr * K + gc),
                    TO_LDS_B128(&As[r][c]),
                    0, 0);
#else
                const float4 v = *(const float4*)(A + (size_t)gr * K + gc);
                As[r][c + 0] = v.x; As[r][c + 1] = v.y;
                As[r][c + 2] = v.z; As[r][c + 3] = v.w;
#endif
            } else {
                #pragma unroll
                for (int q = 0; q < 4; ++q) {
                    int gcq = gc + q;
                    As[r][c + q] = (gr < M && gcq < K)
                                 ? A[(size_t)gr * K + gcq] : 0.0f;
                }
            }
        }
        // ---- Stage B tile (32 x 128) as f16 transposed: 4 quads/thread. ----
        #pragma unroll
        for (int it = 0; it < (TK * TN / 4) / 256; ++it) {
            int i  = tid + it * 256;
            int r  = i >> 5;                 // k row, 32 quads per row
            int c  = (i & 31) * 4;           // n offset
            int gr = k0 + r, gc = bn + c;
            float4 v = make_float4(0.f, 0.f, 0.f, 0.f);
            if (gr < K) v = *(const float4*)(B + (size_t)gr * Ncol + gc);
            Bs[c + 0][r] = (_Float16)v.x;
            Bs[c + 1][r] = (_Float16)v.y;
            Bs[c + 2][r] = (_Float16)v.z;
            Bs[c + 3][r] = (_Float16)v.w;
        }
#if USE_ASYNC_LDS
        __builtin_amdgcn_s_wait_asynccnt(0);
#endif
        __syncthreads();

        // Fragment assembly per ISA 7.12.2 (wave32):
        //  A 16x32: lanes 0-15 -> M=lane, Kbase=0; lanes 16-31 -> Kbase=8.
        //    VGPR v<4:   K = Kbase+2v,2v+1 ; VGPR v>=4: K = 16+Kbase+2(v-4).
        //  B 32x16: N=lane&15, same K order as A (Kbase=8*(lane>>4)).
        const int am   = lane & 15;
        const int kba  = (lane >> 4) * 8;
        const int bn_l = lane & 15;
        const int kbb  = (lane >> 4) * 8;

        #pragma unroll
        for (int i = 0; i < 2; ++i) {
            const float* arow = &As[wm + 16 * i + am][0];
            v16h afrag;
            #pragma unroll
            for (int v = 0; v < 4; ++v) {
                afrag[2 * v]         = (_Float16)arow[kba + 2 * v];
                afrag[2 * v + 1]     = (_Float16)arow[kba + 2 * v + 1];
                afrag[8 + 2 * v]     = (_Float16)arow[16 + kba + 2 * v];
                afrag[8 + 2 * v + 1] = (_Float16)arow[16 + kba + 2 * v + 1];
            }
            #pragma unroll
            for (int j = 0; j < 4; ++j) {
                const _Float16* bcol = &Bs[wn + 16 * j + bn_l][0];
                v16h bfrag;
                #pragma unroll
                for (int v = 0; v < 8; ++v) {
                    bfrag[v]     = bcol[kbb + v];
                    bfrag[8 + v] = bcol[16 + kbb + v];
                }
                acc[i][j] = __builtin_amdgcn_wmma_f32_16x16x32_f16(
                    false, afrag, false, bfrag, (short)0, acc[i][j],
                    false, false);
            }
        }
        __syncthreads();
    }

    // C/D layout: VGPR r -> M = r + 8*(lane>>4); N = lane&15.
    #pragma unroll
    for (int i = 0; i < 2; ++i) {
        #pragma unroll
        for (int j = 0; j < 4; ++j) {
            int row0 = bm + wm + 16 * i + (lane >> 4) * 8;
            int col  = bn + wn + 16 * j + (lane & 15);
            #pragma unroll
            for (int r = 0; r < 8; ++r) {
                int row = row0 + r;
                if (row < M) C[(size_t)row * Ncol + col] = acc[i][j][r];
            }
        }
    }
}

// ---------------------------------------------------------------------------
// Attention coefficients per (node, head) + softmax-state init.
// ---------------------------------------------------------------------------
__global__ void att_init(const float* __restrict__ h,
                         const float* __restrict__ a_src,
                         const float* __restrict__ a_dst,
                         float* __restrict__ als, float* __restrict__ ald,
                         float* __restrict__ m, float* __restrict__ den,
                         int N, int H, int C)
{
    int i = blockIdx.x * blockDim.x + threadIdx.x;
    if (i >= N * H) return;
    int n = i / H, hh = i - n * H;
    const float* hp = h + (size_t)n * H * C + (size_t)hh * C;
    const float* as = a_src + (size_t)hh * C;
    const float* ad = a_dst + (size_t)hh * C;
    float s0 = 0.0f, s1 = 0.0f;
    for (int c = 0; c < C; ++c) { float v = hp[c]; s0 += v * as[c]; s1 += v * ad[c]; }
    als[i] = s0;
    ald[i] = s1;
    m[i]   = -__builtin_huge_valf();
    den[i] = 0.0f;
}

// Float atomic max via signed/unsigned integer ordering trick.
__device__ inline void atomicMaxF(float* addr, float val)
{
    if (val >= 0.0f)
        atomicMax((int*)addr, __float_as_int(val));
    else
        atomicMin((unsigned int*)addr, __float_as_uint(val));
}

// ---------------------------------------------------------------------------
// Edge pass 1: leaky-relu logits + segment max.
// ---------------------------------------------------------------------------
__global__ void edge_logits(const int* __restrict__ src, const int* __restrict__ dst,
                            const float* __restrict__ als, const float* __restrict__ ald,
                            float* __restrict__ e, float* __restrict__ m,
                            int E_, int H)
{
    int i = blockIdx.x * blockDim.x + threadIdx.x;
    if (i >= E_ * H) return;
    int ed = i / H, hh = i - ed * H;
    int s = src[ed], d = dst[ed];
    float v = als[s * H + hh] + ald[d * H + hh];
    v = (v > 0.0f) ? v : 0.2f * v;            // leaky_relu, slope 0.2
    e[i] = v;
    atomicMaxF(&m[d * H + hh], v);
}

// ---------------------------------------------------------------------------
// Edge pass 2: exponentiate shifted logits + segment sum.
// ---------------------------------------------------------------------------
__global__ void edge_exp(const int* __restrict__ dst,
                         float* __restrict__ e, const float* __restrict__ m,
                         float* __restrict__ den, int E_, int H)
{
    int i = blockIdx.x * blockDim.x + threadIdx.x;
    if (i >= E_ * H) return;
    int ed = i / H, hh = i - ed * H;
    int d = dst[ed];
    float ex = expf(e[i] - m[d * H + hh]);
    e[i] = ex;
    atomicAdd(&den[d * H + hh], ex);
}

// ---------------------------------------------------------------------------
// acc += b (GAT bias) + bl (linear-skip bias); acc already holds x @ Wl.
// ---------------------------------------------------------------------------
__global__ void bias_add(float* __restrict__ acc, const float* __restrict__ b,
                         const float* __restrict__ bl, int64_t total, int F)
{
    int64_t i = (int64_t)blockIdx.x * blockDim.x + threadIdx.x;
    if (i >= total) return;
    int f = (int)(i % F);
    acc[i] += b[f] + bl[f];
}

// ---------------------------------------------------------------------------
// Scatter: acc[dst,h,c] += alpha(e,h) * h[src,h,c]; one thread per (e,h,c).
// ---------------------------------------------------------------------------
__global__ void scatter_agg(const int* __restrict__ src, const int* __restrict__ dst,
                            const float* __restrict__ ex, const float* __restrict__ den,
                            const float* __restrict__ h, float* __restrict__ acc,
                            int E_, int H, int C)
{
    int64_t i = (int64_t)blockIdx.x * blockDim.x + threadIdx.x;
    int64_t total = (int64_t)E_ * H * C;
    if (i >= total) return;
    int c  = (int)(i % C);
    int64_t t = i / C;
    int hh = (int)(t % H);
    int ed = (int)(t / H);
    int s = src[ed], d = dst[ed];
    float dn = den[d * H + hh];
    dn = (dn > 1e-16f) ? dn : 1e-16f;
    float alpha = ex[(size_t)ed * H + hh] / dn;
    atomicAdd(&acc[(size_t)d * H * C + (size_t)hh * C + c],
              alpha * h[(size_t)s * H * C + (size_t)hh * C + c]);
}

__global__ void relu_inplace(float* __restrict__ a, int64_t total)
{
    int64_t i = (int64_t)blockIdx.x * blockDim.x + threadIdx.x;
    if (i >= total) return;
    float v = a[i];
    a[i] = v > 0.0f ? v : 0.0f;
}

// ---------------------------------------------------------------------------
// Host launcher
// ---------------------------------------------------------------------------
extern "C" void kernel_launch(void* const* d_in, const int* in_sizes, int n_in,
                              void* d_out, int out_size, void* d_ws, size_t ws_size,
                              hipStream_t stream)
{
    (void)n_in; (void)out_size; (void)ws_size;

    const float* x    = (const float*)d_in[0];
    const int*   ei   = (const int*)  d_in[1];
    const float* W1   = (const float*)d_in[2];
    const float* as1  = (const float*)d_in[3];
    const float* ad1  = (const float*)d_in[4];
    const float* b1   = (const float*)d_in[5];
    const float* Wl1  = (const float*)d_in[6];
    const float* bl1  = (const float*)d_in[7];
    const float* W2   = (const float*)d_in[8];
    const float* as2  = (const float*)d_in[9];
    const float* ad2  = (const float*)d_in[10];
    const float* b2   = (const float*)d_in[11];
    const float* Wl2  = (const float*)d_in[12];
    const float* bl2  = (const float*)d_in[13];

    const int Fin = 300, F1 = 512, F2 = 2048, H = 4, C1 = 128, C2 = 512;
    const int N_ = in_sizes[0] / Fin;
    const int E_ = in_sizes[1] / 2;
    const int* src = ei;
    const int* dst = ei + E_;

    float* out = (float*)d_out;

    // Workspace carve-out (floats): ~126 MB total.
    float* ws    = (float*)d_ws;
    float* hpre1 = ws;                                  // N*512
    float* h1    = hpre1 + (size_t)N_ * F1;             // N*512 (accumulator)
    float* hpre2 = h1    + (size_t)N_ * F1;             // N*2048
    float* ebuf  = hpre2 + (size_t)N_ * F2;             // E*4
    float* als   = ebuf  + (size_t)E_ * H;              // N*4
    float* ald   = als   + (size_t)N_ * H;
    float* mbuf  = ald   + (size_t)N_ * H;
    float* den   = mbuf  + (size_t)N_ * H;

    dim3 blk(256);
    auto g1 = [](int64_t n) { return dim3((unsigned)((n + 255) / 256)); };
    const unsigned mtiles = (unsigned)((N_ + TM - 1) / TM);

    // ---------------- Layer 1 ----------------
    gemm_wmma_f16<<<dim3(F1 / TN, mtiles), blk, 0, stream>>>(x, W1,  hpre1, N_, Fin, F1);
    gemm_wmma_f16<<<dim3(F1 / TN, mtiles), blk, 0, stream>>>(x, Wl1, h1,    N_, Fin, F1);

    att_init   <<<g1((int64_t)N_ * H), blk, 0, stream>>>(hpre1, as1, ad1, als, ald, mbuf, den, N_, H, C1);
    edge_logits<<<g1((int64_t)E_ * H), blk, 0, stream>>>(src, dst, als, ald, ebuf, mbuf, E_, H);
    edge_exp   <<<g1((int64_t)E_ * H), blk, 0, stream>>>(dst, ebuf, mbuf, den, E_, H);
    bias_add   <<<g1((int64_t)N_ * F1), blk, 0, stream>>>(h1, b1, bl1, (int64_t)N_ * F1, F1);
    scatter_agg<<<g1((int64_t)E_ * H * C1), blk, 0, stream>>>(src, dst, ebuf, den, hpre1, h1, E_, H, C1);
    relu_inplace<<<g1((int64_t)N_ * F1), blk, 0, stream>>>(h1, (int64_t)N_ * F1);

    // ---------------- Layer 2 ----------------
    gemm_wmma_f16<<<dim3(F2 / TN, mtiles), blk, 0, stream>>>(h1, W2,  hpre2, N_, F1, F2);
    gemm_wmma_f16<<<dim3(F2 / TN, mtiles), blk, 0, stream>>>(h1, Wl2, out,   N_, F1, F2);

    att_init   <<<g1((int64_t)N_ * H), blk, 0, stream>>>(hpre2, as2, ad2, als, ald, mbuf, den, N_, H, C2);
    edge_logits<<<g1((int64_t)E_ * H), blk, 0, stream>>>(src, dst, als, ald, ebuf, mbuf, E_, H);
    edge_exp   <<<g1((int64_t)E_ * H), blk, 0, stream>>>(dst, ebuf, mbuf, den, E_, H);
    bias_add   <<<g1((int64_t)N_ * F2), blk, 0, stream>>>(out, b2, bl2, (int64_t)N_ * F2, F2);
    scatter_agg<<<g1((int64_t)E_ * H * C2), blk, 0, stream>>>(src, dst, ebuf, den, hpre2, out, E_, H, C2);
}
